// HybridHyperedgeGenerator_90323162235048
// MI455X (gfx1250) — matrix operand order repaired and checked by reading, not compiled, not run on hardware
//
#include <hip/hip_runtime.h>
#include <hip/hip_bf16.h>

#define HID 128

typedef __attribute__((ext_vector_type(16))) _Float16 v16h;
typedef __attribute__((ext_vector_type(8)))  float    v8f;

union FragA { v16h v; unsigned int u[8]; _Float16 h[16]; };
union FragB { v16h v; uint4 q[2]; };

static __device__ inline v8f wmma32f16(v16h a, v16h b, v8f c) {
    // D = A(16x32 f16) * B(32x16 f16) + C(16x16 f32)
    return __builtin_amdgcn_wmma_f32_16x16x32_f16(false, a, false, b, (short)0, c,
                                                  false, false);
}

// Wave-private LDS ordering: DS ops from one wave complete in order (ISA ch.11);
// a dscnt wait + compiler memory fence replaces a full workgroup barrier.
static __device__ inline void lds_fence_wave() {
    asm volatile("s_wait_dscnt 0x0" ::: "memory");
}

// Hard scheduling fence: nothing crosses (prevents load-sinking to first use).
static __device__ inline void sched_fence() { __builtin_amdgcn_sched_barrier(0); }

// ---------------------------------------------------------------------------
// Weight convert + transpose: wt[c*K + k] = (f16) w[k*128 + c]
// ---------------------------------------------------------------------------
__global__ __launch_bounds__(256) void convT(const float* __restrict__ w,
                                             _Float16* __restrict__ wt, int K) {
    int t = blockIdx.x * blockDim.x + threadIdx.x;
    if (t >= K * 128) return;
    int k = t >> 7, c = t & 127;
    wt[(size_t)c * K + k] = (_Float16)w[t];
}

// v[k] = mean_j wf[k][j];  scal = {softmax(attn)[0], softmax(attn)[1], mean(bf)}
__global__ __launch_bounds__(128) void prep_scalars(const float* __restrict__ wf,
                                                    const float* __restrict__ bf,
                                                    const float* __restrict__ attn,
                                                    float* __restrict__ vvec,
                                                    float* __restrict__ scal) {
    int k = threadIdx.x;
    float s = 0.f;
    for (int j = 0; j < 128; ++j) s += wf[k * 128 + j];
    vvec[k] = s * (1.f / 128.f);
    if (k == 0) {
        float a0 = attn[0], a1 = attn[1];
        float m = fmaxf(a0, a1);
        float e0 = __expf(a0 - m), e1 = __expf(a1 - m);
        float inv = 1.f / (e0 + e1);
        scal[0] = e0 * inv;
        scal[1] = e1 * inv;
        float sb = 0.f;
        for (int j = 0; j < 128; ++j) sb += bf[j];
        scal[2] = sb * (1.f / 128.f);
    }
}

// ---------------------------------------------------------------------------
// WMMA GEMM: C[N x 128] = A[N x K] @ Wt^T + bias, with fused epilogues.
//   Wt is pre-transposed f16 [128 cols][K].
//   Block = 256 threads (8 waves); each wave: 16 rows x 128 cols; grid = N/128.
//   EPI 0: relu -> f16        EPI 1: outF = alpha*(acc+bias) (+= old if beta)
//   EPI 2: row-L2-normalize(acc+bias) -> f16
// ---------------------------------------------------------------------------
template <bool A_F16, int EPI>
__global__ __launch_bounds__(256) void gemm_nt(const void* __restrict__ Av, int K,
                                               const _Float16* __restrict__ Wt,
                                               const float* __restrict__ bias,
                                               float* __restrict__ outF,
                                               _Float16* __restrict__ outH,
                                               const float* __restrict__ alphaPtr,
                                               int betaFlag) {
    const int wave = threadIdx.x >> 5;
    const int lane = threadIdx.x & 31;
    const int lan15 = lane & 15;
    const int hi = (lane >> 4) & 1;
    const int rowTile = blockIdx.x * 128 + wave * 16;
    const int arow = rowTile + lan15;

    v8f acc[8];
#pragma unroll
    for (int t = 0; t < 8; ++t)
#pragma unroll
        for (int r = 0; r < 8; ++r) acc[t][r] = 0.f;

    const int nk = K >> 5;
    for (int c = 0; c < nk; ++c) {
        // A fragment: 16x32 f16 layout (lane<16: K base 0, lane>=16: K base 8)
        FragA a;
        const int kb = c * 32 + (hi ? 8 : 0);
        if (A_F16) {
            const _Float16* A = (const _Float16*)Av;
#pragma unroll
            for (int v = 0; v < 8; ++v) {
                int k = kb + ((v & 4) << 2) + ((v & 3) << 1);
                a.u[v] = *(const unsigned int*)(A + (size_t)arow * K + k);
            }
        } else {
            const float* A = (const float*)Av;
#pragma unroll
            for (int v = 0; v < 8; ++v) {
                int k = kb + ((v & 4) << 2) + ((v & 3) << 1);
                float2 t2 = *(const float2*)(A + (size_t)arow * K + k);
                a.h[2 * v]     = (_Float16)t2.x;
                a.h[2 * v + 1] = (_Float16)t2.y;
            }
        }
        // Preload ALL 8 B fragments; sched barrier keeps the loads grouped
        // (one clause, staggered waits) instead of sunk to first use.
        const int kk = c * 32 + (hi ? 8 : 0);
        FragB b[8];
#pragma unroll
        for (int ct = 0; ct < 8; ++ct) {
            const _Float16* src = Wt + (size_t)(ct * 16 + lan15) * K + kk;
            b[ct].q[0] = *(const uint4*)(src);
            b[ct].q[1] = *(const uint4*)(src + 16);
        }
        sched_fence();
#pragma unroll
        for (int ct = 0; ct < 8; ++ct) acc[ct] = wmma32f16(a.v, b[ct].v, acc[ct]);
    }

    const float alpha = alphaPtr ? alphaPtr[0] : 1.0f;
    const int rbase = rowTile + (hi ? 8 : 0);

    if (EPI == 1) {
#pragma unroll
        for (int ct = 0; ct < 8; ++ct) {
            const int col = ct * 16 + lan15;
            const float bc = bias[col];
#pragma unroll
            for (int r = 0; r < 8; ++r) {
                size_t idx = (size_t)(rbase + r) * HID + col;
                float o = alpha * (acc[ct][r] + bc);
                if (betaFlag) o += outF[idx];
                outF[idx] = o;
            }
        }
    } else if (EPI == 0) {
#pragma unroll
        for (int ct = 0; ct < 8; ++ct) {
            const int col = ct * 16 + lan15;
            const float bc = bias[col];
#pragma unroll
            for (int r = 0; r < 8; ++r) {
                float v2 = fmaxf(acc[ct][r] + bc, 0.f);
                outH[(size_t)(rbase + r) * HID + col] = (_Float16)v2;
            }
        }
    } else {
        // add bias, then row norm across the wave's full 128 columns
#pragma unroll
        for (int ct = 0; ct < 8; ++ct) {
            const float bc = bias[ct * 16 + lan15];
#pragma unroll
            for (int r = 0; r < 8; ++r) acc[ct][r] += bc;
        }
        float ss[8];
#pragma unroll
        for (int r = 0; r < 8; ++r) {
            float s = 0.f;
#pragma unroll
            for (int ct = 0; ct < 8; ++ct) s += acc[ct][r] * acc[ct][r];
            ss[r] = s;
        }
        // xor masks 1/2/4/8 stay inside each 16-lane half (matches C layout)
#pragma unroll
        for (int m = 1; m <= 8; m <<= 1)
#pragma unroll
            for (int r = 0; r < 8; ++r) ss[r] += __shfl_xor(ss[r], m, 32);
#pragma unroll
        for (int r = 0; r < 8; ++r) {
            const float sc = 1.f / fmaxf(sqrtf(ss[r]), 1e-12f);
#pragma unroll
            for (int ct = 0; ct < 8; ++ct) {
                outH[(size_t)(rbase + r) * HID + ct * 16 + lan15] =
                    (_Float16)(acc[ct][r] * sc);
            }
        }
    }
}

// ---------------------------------------------------------------------------
// Fused sim GEMM + streaming top-8 (self excluded; equals top-9 drop self).
// Block = 256 threads (8 waves); wave owns 32 query rows held as A fragments.
// 32 columns (2 WMMA col-tiles) per iteration. Software-pipelined: the B
// registers are refilled for the NEXT pair right after the WMMAs consume
// them, so the loads fly during the LDS transpose + top-k scan phase.
// ---------------------------------------------------------------------------
#define SPAD 36  // row pad in floats: 36*4 = 144B keeps float4 alignment

__global__ __launch_bounds__(256) void simtopk(const _Float16* __restrict__ nh,
                                               int* __restrict__ edges, int N) {
    __shared__ float smem[8][32][SPAD];
    const int wave = threadIdx.x >> 5;
    const int lane = threadIdx.x & 31;
    const int lan15 = lane & 15;
    const int hi = (lane >> 4) & 1;
    const int qbase = blockIdx.x * 256 + wave * 32;
    const int myrow = qbase + lane;
    const int kkbase = hi ? 8 : 0;

    FragA af[2][4];
#pragma unroll
    for (int t = 0; t < 2; ++t)
#pragma unroll
        for (int c = 0; c < 4; ++c) {
            const int row = qbase + t * 16 + lan15;
            const int kb = c * 32 + (hi ? 8 : 0);
#pragma unroll
            for (int v = 0; v < 8; ++v) {
                int k = kb + ((v & 4) << 2) + ((v & 3) << 1);
                af[t][c].u[v] = *(const unsigned int*)(nh + (size_t)row * HID + k);
            }
        }

    float tv[8];
    int ti[8];
#pragma unroll
    for (int k = 0; k < 8; ++k) { tv[k] = -2.0f; ti[k] = -1; }

    FragB b[2][4];
    auto loadB = [&](int cbase) {
#pragma unroll
        for (int s = 0; s < 2; ++s)
#pragma unroll
            for (int c = 0; c < 4; ++c) {
                const _Float16* src =
                    nh + (size_t)(cbase + s * 16 + lan15) * HID + c * 32 + kkbase;
                b[s][c].q[0] = *(const uint4*)(src);
                b[s][c].q[1] = *(const uint4*)(src + 16);
            }
    };

    const int npairs = N >> 5;  // 32 columns per iteration
    loadB(0);
    for (int pair = 0; pair < npairs; ++pair) {
        const int cb = pair << 5;

        v8f acc[2][2];  // [rowTile][colTile]
#pragma unroll
        for (int t = 0; t < 2; ++t)
#pragma unroll
            for (int s = 0; s < 2; ++s)
#pragma unroll
                for (int r = 0; r < 8; ++r) acc[t][s][r] = 0.f;
#pragma unroll
        for (int c = 0; c < 4; ++c)
#pragma unroll
            for (int t = 0; t < 2; ++t)
#pragma unroll
                for (int s = 0; s < 2; ++s)
                    acc[t][s] = wmma32f16(af[t][c].v, b[s][c].v, acc[t][s]);

        // Refill B for the next pair NOW (registers are dead post-WMMA); the
        // loads overlap the transpose + scan below. Clamped so last iteration
        // issues harmless in-bounds re-loads.
        {
            const int nb = (pair + 1 < npairs ? pair + 1 : pair) << 5;
            loadB(nb);
            if (pair + 2 < npairs) {
                __builtin_prefetch(nh + (size_t)(cb + 64 + lan15) * HID + kkbase, 0, 1);
                __builtin_prefetch(nh + (size_t)(cb + 80 + lan15) * HID + kkbase, 0, 1);
            }
        }

        // Transpose 32x32 sim block through wave-private LDS.
        const int r0 = hi * 8;
#pragma unroll
        for (int t = 0; t < 2; ++t)
#pragma unroll
            for (int s = 0; s < 2; ++s)
#pragma unroll
                for (int r = 0; r < 8; ++r)
                    smem[wave][t * 16 + r + r0][s * 16 + lan15] = acc[t][s][r];
        lds_fence_wave();

        // Streaming top-8 insertion; lane owns row qbase+lane.
        const float4* rowp = (const float4*)&smem[wave][lane][0];
#pragma unroll
        for (int j4 = 0; j4 < 8; ++j4) {
            const float4 vv = rowp[j4];
            const float vals[4] = {vv.x, vv.y, vv.z, vv.w};
#pragma unroll
            for (int e = 0; e < 4; ++e) {
                const int col = cb + j4 * 4 + e;
                const float v = vals[e];
                if (col != myrow && v > tv[7]) {
                    float cv = v; int ci = col;
#pragma unroll
                    for (int k = 0; k < 8; ++k) {
                        if (cv > tv[k]) {
                            float t1 = tv[k]; tv[k] = cv; cv = t1;
                            int   t2 = ti[k]; ti[k] = ci; ci = t2;
                        }
                    }
                }
            }
        }
        lds_fence_wave();  // loads consumed before next iteration's stores
    }

    int* ep = edges + (size_t)myrow * 9;
    ep[0] = myrow;
#pragma unroll
    for (int k = 0; k < 8; ++k) ep[1 + k] = ti[k];
}

// p[i] = dot(fused_pre[i], vvec)
__global__ __launch_bounds__(256) void compute_p(const float* __restrict__ fp,
                                                 const float* __restrict__ vvec,
                                                 float* __restrict__ p, int N) {
    int i = blockIdx.x * blockDim.x + threadIdx.x;
    if (i >= N) return;
    const float* row = fp + (size_t)i * HID;
    float s = 0.f;
#pragma unroll 8
    for (int k = 0; k < HID; ++k) s += row[k] * vvec[k];
    p[i] = s;
}

// w[i] = sigmoid((sum_e p[edges[i,e]])/9 + c); keep/zero per threshold
__global__ __launch_bounds__(256) void score_w(const int* __restrict__ edges,
                                               const float* __restrict__ p,
                                               const float* __restrict__ scal,
                                               float* __restrict__ wout,
                                               int* __restrict__ keep, int N) {
    int i = blockIdx.x * blockDim.x + threadIdx.x;
    if (i >= N) return;
    const int* ep = edges + (size_t)i * 9;
    float s = 0.f;
#pragma unroll
    for (int e = 0; e < 9; ++e) s += p[ep[e]];
    float val = s * (1.f / 9.f) + scal[2];
    float w = 1.f / (1.f + __expf(-val));
    int kp = (w > 0.5f) ? 1 : 0;
    wout[i] = kp ? w : 0.f;
    keep[i] = kp;
}

__global__ __launch_bounds__(256) void zero4(float4* __restrict__ q, size_t n4) {
    size_t i = (size_t)blockIdx.x * blockDim.x + threadIdx.x;
    size_t stride = (size_t)gridDim.x * blockDim.x;
    float4 z; z.x = 0.f; z.y = 0.f; z.z = 0.f; z.w = 0.f;
    for (; i < n4; i += stride) q[i] = z;
}

__global__ __launch_bounds__(256) void scatterH(const int* __restrict__ edges,
                                                const int* __restrict__ keep,
                                                float* __restrict__ H, int N) {
    int t = blockIdx.x * blockDim.x + threadIdx.x;
    if (t >= N * 9) return;
    int i = t / 9;
    if (keep[i]) H[(size_t)edges[t] * N + i] = 1.0f;
}

// ---------------------------------------------------------------------------
extern "C" void kernel_launch(void* const* d_in, const int* in_sizes, int n_in,
                              void* d_out, int out_size, void* d_ws, size_t ws_size,
                              hipStream_t stream) {
    const int N = in_sizes[0] / 512;  // x0 is [N,512]
    const int D0 = 512;

    const float* x0   = (const float*)d_in[0];
    const float* x1   = (const float*)d_in[1];
    const float* w1_0 = (const float*)d_in[2];
    const float* b1_0 = (const float*)d_in[3];
    const float* w2_0 = (const float*)d_in[4];
    const float* b2_0 = (const float*)d_in[5];
    const float* w1_1 = (const float*)d_in[6];
    const float* b1_1 = (const float*)d_in[7];
    const float* w2_1 = (const float*)d_in[8];
    const float* b2_1 = (const float*)d_in[9];
    const float* attn = (const float*)d_in[10];
    const float* wf   = (const float*)d_in[11];
    const float* bf   = (const float*)d_in[12];
    float* Hout = (float*)d_out;
    float* wout = Hout + (size_t)N * N;

    // workspace carving (256B aligned)
    char* ws = (char*)d_ws;
    size_t off = 0;
    auto carve = [&](size_t bytes) -> void* {
        void* p = ws + off;
        off = (off + bytes + 255) & ~(size_t)255;
        return p;
    };
    _Float16* wt1_0 = (_Float16*)carve((size_t)D0 * HID * 2);
    _Float16* wt1_1 = (_Float16*)carve((size_t)HID * HID * 2);
    _Float16* wt2_0 = (_Float16*)carve((size_t)HID * HID * 2);
    _Float16* wt2_1 = (_Float16*)carve((size_t)HID * HID * 2);
    _Float16* wtf   = (_Float16*)carve((size_t)HID * HID * 2);
    _Float16* h0    = (_Float16*)carve((size_t)N * HID * 2);
    _Float16* h1    = (_Float16*)carve((size_t)N * HID * 2);
    float*    fpre  = (float*)carve((size_t)N * HID * 4);
    _Float16* normd = (_Float16*)carve((size_t)N * HID * 2);
    float*    pbuf  = (float*)carve((size_t)N * 4);
    int*      edges = (int*)carve((size_t)N * 9 * 4);
    float*    vvec  = (float*)carve(HID * 4);
    float*    scal  = (float*)carve(4 * 4);
    int*      keep  = (int*)carve((size_t)N * 4);

    // 1) zero-fill H (bandwidth-bound floor)
    zero4<<<8192, 256, 0, stream>>>((float4*)Hout, (size_t)N * N / 4);

    // 2) weight conversion / transposition + scalar prep
    convT<<<(D0 * 128 + 255) / 256, 256, 0, stream>>>(w1_0, wt1_0, D0);
    convT<<<(HID * 128 + 255) / 256, 256, 0, stream>>>(w1_1, wt1_1, HID);
    convT<<<(HID * 128 + 255) / 256, 256, 0, stream>>>(w2_0, wt2_0, HID);
    convT<<<(HID * 128 + 255) / 256, 256, 0, stream>>>(w2_1, wt2_1, HID);
    convT<<<(HID * 128 + 255) / 256, 256, 0, stream>>>(wf, wtf, HID);
    prep_scalars<<<1, 128, 0, stream>>>(wf, bf, attn, vvec, scal);

    // 3) encode MLPs (WMMA), fused attention combine, projection + normalize
    gemm_nt<false, 0><<<N / 128, 256, 0, stream>>>(x0, D0, wt1_0, b1_0, nullptr, h0,
                                                   nullptr, 0);
    gemm_nt<false, 0><<<N / 128, 256, 0, stream>>>(x1, HID, wt1_1, b1_1, nullptr, h1,
                                                   nullptr, 0);
    gemm_nt<true, 1><<<N / 128, 256, 0, stream>>>(h0, HID, wt2_0, b2_0, fpre, nullptr,
                                                  scal + 0, 0);
    gemm_nt<true, 1><<<N / 128, 256, 0, stream>>>(h1, HID, wt2_1, b2_1, fpre, nullptr,
                                                  scal + 1, 1);
    gemm_nt<false, 2><<<N / 128, 256, 0, stream>>>(fpre, HID, wtf, bf, nullptr, normd,
                                                   nullptr, 0);

    // 4) fused cosine-sim GEMM + streaming per-row top-8 (dominant compute)
    simtopk<<<N / 256, 256, 0, stream>>>(normd, edges, N);

    // 5) collapsed hyperedge scoring
    compute_p<<<N / 256, 256, 0, stream>>>(fpre, vvec, pbuf, N);
    score_w<<<N / 256, 256, 0, stream>>>(edges, pbuf, scal, wout, keep, N);

    // 6) incidence scatter
    scatterH<<<(N * 9 + 255) / 256, 256, 0, stream>>>(edges, keep, Hout, N);
}
